// FastSpeech2_31112743092730
// MI455X (gfx1250) — matrix operand order repaired and checked by reading, not compiled, not run on hardware
//
#include <hip/hip_runtime.h>
#include <hip/hip_bf16.h>

// ---------------- model constants ----------------
#define BB   32
#define SS   240
#define DD   256
#define HH   2
#define HDD  128
#define LLY  4
#define MELN 960
#define MBN  80

typedef _Float16 v16h __attribute__((ext_vector_type(16)));
typedef _Float16 v8h  __attribute__((ext_vector_type(8)));
typedef float    v8f  __attribute__((ext_vector_type(8)));
typedef unsigned int u32x4 __attribute__((ext_vector_type(4)));
typedef int i32x4 __attribute__((ext_vector_type(4)));
typedef int i32x8 __attribute__((ext_vector_type(8)));
typedef __attribute__((address_space(3))) _Float16 lf16;

#if defined(__gfx1250__) && __has_builtin(__builtin_amdgcn_tensor_load_to_lds) && \
    __has_builtin(__builtin_amdgcn_s_wait_tensorcnt)
#define HAVE_TDM 1
#else
#define HAVE_TDM 0
#endif

__device__ __forceinline__ v8f wmma_f16(v16h a, v16h b, v8f c) {
  // D = A(16x32 f16) * B(32x16 f16) + C(16x16 f32)
  return __builtin_amdgcn_wmma_f32_16x16x32_f16(false, a, false, b, (short)0, c,
                                                false, false);
}

#if HAVE_TDM
// Issue a TDM load of a 32x32 f16 tile (row stride K elements) into LDS.
// LDS rows are padded by 1 DWORD per 64B (bank de-conflict): stride 34 halfs.
__device__ __forceinline__ void tdm_load_a(const _Float16* gsrc, unsigned lds_off,
                                           int K) {
  unsigned long long ga = (unsigned long long)gsrc;
  u32x4 g0;
  g0[0] = 1u;                                   // count=1, user descriptor
  g0[1] = lds_off;                              // LDS byte address
  g0[2] = (unsigned)(ga & 0xffffffffu);         // global_addr[31:0]
  g0[3] = (unsigned)((ga >> 32) & 0x01ffffffu)  // global_addr[56:32]
          | (2u << 30);                         // type = 2 ("image")
  i32x8 g1;
  g1[0] = 0x10000              // data_size = 1 (2 bytes)
          | (1 << 20)          // pad_enable
          | (3 << 22);         // pad_interval: every 16 DWORDs (64B row)
                               // pad_amount = 0 -> 1 DWORD
  g1[1] = (32 << 16);          // tensor_dim0[15:0] = 32 (bits 63:48)
  g1[2] = (32 << 16);          // tensor_dim1[15:0] = 32 (bits 95:80)
  g1[3] = (32 << 16);          // tile_dim0 = 32 (bits 127:112)
  g1[4] = 32;                  // tile_dim1 = 32 (tile_dim2 = 0)
  g1[5] = K;                   // tensor_dim0_stride[31:0] (elements)
  g1[6] = 0;
  g1[7] = 0;
  i32x4 z4 = {0, 0, 0, 0};
#if __clang_major__ >= 23
  i32x8 z8 = {0, 0, 0, 0, 0, 0, 0, 0};
  __builtin_amdgcn_tensor_load_to_lds(g0, g1, z4, z4, z8, 0);
#else
  __builtin_amdgcn_tensor_load_to_lds(g0, g1, z4, z4, 0);
#endif
}
#endif

// ---------------- generic WMMA GEMM: C[M,N] = A[M,K] * W[N,K]^T + bias ----------------
// One wave computes 2 x NT tiles of 16x16. K % 32 == 0, N % (NT*16) == 0, M % 32 == 0.
// A tiles are staged to LDS by the Tensor Data Mover (double-buffered).
template <int MT, int NT, int RELU, int OUT16>
__global__ void __launch_bounds__(128) gemm_k(
    const _Float16* __restrict__ A, const _Float16* __restrict__ W,
    const float* __restrict__ bias, float* __restrict__ Cf,
    _Float16* __restrict__ Ch, int M, int N, int K) {
  const int lane = threadIdx.x & 31;
  const int wid  = blockIdx.x * (blockDim.x >> 5) + (threadIdx.x >> 5);
  const int nblk = N / (NT * 16);
  const int mb = wid / nblk, nb = wid % nblk;
  const int m0 = mb * MT * 16;
  if (m0 >= M) return;
  const int n0 = nb * NT * 16;
  const int hf = lane >> 4, l16 = lane & 15;

  v8f acc[MT][NT];
#pragma unroll
  for (int nt = 0; nt < NT; nt++) {
    float bv = bias[n0 + nt * 16 + l16];
#pragma unroll
    for (int mt = 0; mt < MT; mt++)
#pragma unroll
      for (int r = 0; r < 8; r++) acc[mt][nt][r] = bv;
  }

#if HAVE_TDM
  // --- TDM-staged A tiles, double buffered: 32 rows x 32 k, LDS row stride 34 ---
  __shared__ _Float16 ldsA[4][2][32 * 34];
  const int w = threadIdx.x >> 5;
  const _Float16* Abase = A + (size_t)m0 * K;
  tdm_load_a(Abase, (unsigned)(unsigned long long)(lf16*)&ldsA[w][0][0], K);
  int buf = 0;
  for (int k0 = 0; k0 < K; k0 += 32) {
    if (k0 + 32 < K) {
      tdm_load_a(Abase + k0 + 32,
                 (unsigned)(unsigned long long)(lf16*)&ldsA[w][buf ^ 1][0], K);
      __builtin_amdgcn_s_wait_tensorcnt(1);
    } else {
      __builtin_amdgcn_s_wait_tensorcnt(0);
    }
    asm volatile("" ::: "memory"); // keep LDS reads below the tensorcnt wait
    v16h af[MT], bf[NT];
#pragma unroll
    for (int mt = 0; mt < MT; mt++) {
      const _Float16* ap = &ldsA[w][buf][(mt * 16 + l16) * 34 + hf * 16];
      union { v16h v; v8h h2[2]; } u;
      u.h2[0] = *(const v8h*)ap;        // k = hf*16 .. +7
      u.h2[1] = *(const v8h*)(ap + 8);  // k = hf*16+8 .. +15
      af[mt] = u.v;
    }
#pragma unroll
    for (int nt = 0; nt < NT; nt++)
      bf[nt] = *(const v16h*)(W + (size_t)(n0 + nt * 16 + l16) * K + k0 + hf * 16);
#pragma unroll
    for (int mt = 0; mt < MT; mt++)
#pragma unroll
      for (int nt = 0; nt < NT; nt++)
        acc[mt][nt] = wmma_f16(af[mt], bf[nt], acc[mt][nt]);
    buf ^= 1;
  }
#else
  for (int k0 = 0; k0 < K; k0 += 32) {
    v16h af[MT], bf[NT];
#pragma unroll
    for (int mt = 0; mt < MT; mt++) {
      const _Float16* ap = A + (size_t)(m0 + mt * 16 + l16) * K + k0 + hf * 16;
      union { v16h v; v8h h2[2]; } u;
      u.h2[0] = *(const v8h*)ap;
      u.h2[1] = *(const v8h*)(ap + 8);
      af[mt] = u.v;
    }
#pragma unroll
    for (int nt = 0; nt < NT; nt++)
      bf[nt] = *(const v16h*)(W + (size_t)(n0 + nt * 16 + l16) * K + k0 + hf * 16);
#pragma unroll
    for (int mt = 0; mt < MT; mt++)
#pragma unroll
      for (int nt = 0; nt < NT; nt++)
        acc[mt][nt] = wmma_f16(af[mt], bf[nt], acc[mt][nt]);
  }
#endif

#pragma unroll
  for (int mt = 0; mt < MT; mt++)
#pragma unroll
    for (int nt = 0; nt < NT; nt++)
#pragma unroll
      for (int r = 0; r < 8; r++) {
        int row = m0 + mt * 16 + hf * 8 + r;
        int col = n0 + nt * 16 + l16;
        float v = acc[mt][nt][r];
        if (RELU) v = v > 0.f ? v : 0.f;
        if (OUT16) Ch[(size_t)row * N + col] = (_Float16)v;
        else       Cf[(size_t)row * N + col] = v;
      }
}

// ---------------- V transpose: vT[b, d, key] = qkv[b*S+key, 512+d], zero-padded tail ----
__global__ void vtrans_k(const _Float16* __restrict__ qkv, _Float16* __restrict__ vT,
                         int S, int Sp) {
  int i = blockIdx.x * blockDim.x + threadIdx.x;
  int total = BB * DD * Sp;
  if (i >= total) return;
  int t = i % Sp;
  int bd = i / Sp;
  int d = bd % DD, b = bd / DD;
  _Float16 v = (_Float16)0.f;
  if (t < S) v = qkv[(size_t)(b * S + t) * (3 * DD) + 2 * DD + d];
  vT[i] = v; // coalesced over t
}

// ---------------- transposed flash attention ----------------
// One wave per (b, h, 16-query tile). Computes S^T = K*Q^T and O^T = V^T*P^T so
// softmax stats are per-lane scalars and every memory op is a vector op. Loads
// are batched ahead of the WMMA chains so latency overlaps VALU/WMMA work.
__global__ void __launch_bounds__(128) attn_k(const _Float16* __restrict__ qkv,
                                              const _Float16* __restrict__ vT,
                                              _Float16* __restrict__ obuf,
                                              int S, int Sp) {
  const int lane = threadIdx.x & 31;
  const int w    = threadIdx.x >> 5;
  const int wid  = blockIdx.x * 4 + w;
  const int qtiles = S >> 4;
  const int qt = wid % qtiles;
  const int bh = wid / qtiles;
  const int h = bh & 1, b = bh >> 1;
  const int hf = lane >> 4, l16 = lane & 15;
  const int ld = 3 * DD;
  const _Float16* qb  = qkv + (size_t)b * S * ld + h * HDD;
  const _Float16* kb  = qb + DD;
  const _Float16* vtb = vT + ((size_t)b * DD + h * HDD) * Sp;
  const float scale = 0.08838834764831845f; // 1/sqrt(128)

  // Q^T B-fragments: 4 chunks of 32 dims; contiguous 32B loads
  v16h qfB[4];
  const int qrow = qt * 16 + l16;
#pragma unroll
  for (int c = 0; c < 4; c++)
    qfB[c] = *(const v16h*)(qb + (size_t)qrow * ld + c * 32 + hf * 16);

  v8f oT[8]; // O^T chunks: dim = nc*16 + hf*8 + r, query = l16
#pragma unroll
  for (int nc = 0; nc < 8; nc++)
#pragma unroll
    for (int r = 0; r < 8; r++) oT[nc][r] = 0.f;
  float mstat = -3.0e38f, lstat = 0.f; // per-lane (per-query) stats

  __shared__ _Float16 pld[4][16][32]; // [wave][query][key-in-block]
  const int ktiles = S >> 4;
  const int kpairs = (ktiles + 1) >> 1;

  for (int kp = 0; kp < kpairs; kp++) {
    const int kb0 = kp * 32;
    if (kp + 1 < kpairs) { // prefetch next block's K rows and V^T rows
      __builtin_prefetch(kb + (size_t)(kb0 + 32 + l16) * ld, 0, 1);
      __builtin_prefetch(vtb + (size_t)l16 * Sp + kb0 + 32, 0, 1);
    }
    // ---- phase 1: batch all K-row fragment loads (2 tiles x 4 chunks) ----
    v16h kf[2][4];
#pragma unroll
    for (int j = 0; j < 2; j++) {
      int kt = kp * 2 + j;
      if (kt < ktiles) {
#pragma unroll
        for (int cd = 0; cd < 4; cd++) {
          const _Float16* kr = kb + (size_t)(kt * 16 + l16) * ld + cd * 32 + hf * 16;
          union { v16h v; v8h h2[2]; } u;
          u.h2[0] = *(const v8h*)kr;
          u.h2[1] = *(const v8h*)(kr + 8);
          kf[j][cd] = u.v;
        }
      }
    }
    // ---- phase 2: S^T WMMA chain (M=key, N=query) ----
    v8f s[2];
#pragma unroll
    for (int j = 0; j < 2; j++) {
      int kt = kp * 2 + j;
      if (kt < ktiles) {
        v8f c;
#pragma unroll
        for (int r = 0; r < 8; r++) c[r] = 0.f;
#pragma unroll
        for (int cd = 0; cd < 4; cd++) c = wmma_f16(kf[j][cd], qfB[cd], c);
#pragma unroll
        for (int r = 0; r < 8; r++) s[j][r] = c[r] * scale;
      } else {
#pragma unroll
        for (int r = 0; r < 8; r++) s[j][r] = -1.0e30f; // masked tail tile
      }
    }
    // ---- phase 3: issue all V^T fragment loads now; latency overlaps softmax ----
    v16h vf[8];
#pragma unroll
    for (int nc = 0; nc < 8; nc++) {
      const _Float16* vr = vtb + (size_t)(nc * 16 + l16) * Sp + kb0 + hf * 16;
      union { v16h v; v8h h2[2]; } u;
      u.h2[0] = *(const v8h*)vr;        // keys kb0+hf*16..+7
      u.h2[1] = *(const v8h*)(vr + 8);  // keys kb0+hf*16+8..+15
      vf[nc] = u.v;
    }
    // ---- phase 4: per-query online softmax (register reduce + one shuffle) ----
    float bm = s[0][0];
#pragma unroll
    for (int r = 1; r < 8; r++) bm = fmaxf(bm, s[0][r]);
#pragma unroll
    for (int r = 0; r < 8; r++) bm = fmaxf(bm, s[1][r]);
    bm = fmaxf(bm, __shfl_xor(bm, 16, 32));
    float mn = fmaxf(mstat, bm);
    float alpha = __expf(mstat - mn);
    v8h ph[2];
    float rs = 0.f;
#pragma unroll
    for (int j = 0; j < 2; j++)
#pragma unroll
      for (int r = 0; r < 8; r++) {
        float pv = __expf(s[j][r] - mn);
        rs += pv;
        ph[j][r] = (_Float16)pv;
      }
    rs += __shfl_xor(rs, 16, 32);
    lstat = lstat * alpha + rs;
    mstat = mn;
#pragma unroll
    for (int nc = 0; nc < 8; nc++)
#pragma unroll
      for (int r = 0; r < 8; r++) oT[nc][r] *= alpha;

    // ---- phase 5: stage P^T through per-wave LDS (HW keeps same-wave LDS in
    // order; only compiler reordering must be fenced) ----
    asm volatile("" ::: "memory");
    *(v8h*)&pld[w][l16][hf * 8]      = ph[0]; // keys hf*8..+7
    *(v8h*)&pld[w][l16][16 + hf * 8] = ph[1]; // keys 16+hf*8..+7
    asm volatile("" ::: "memory");
    v16h pf = *(const v16h*)&pld[w][l16][hf * 16]; // B-frag: key=hf*16+e, query=l16

    // ---- phase 6: O^T += V^T * P^T, back-to-back WMMAs ----
#pragma unroll
    for (int nc = 0; nc < 8; nc++) oT[nc] = wmma_f16(vf[nc], pf, oT[nc]);
  }
  // normalize and store: contiguous 16B stores per chunk
  float inv = 1.0f / lstat;
  const int tok = b * S + qt * 16 + l16;
#pragma unroll
  for (int nc = 0; nc < 8; nc++) {
    v8h hv;
#pragma unroll
    for (int r = 0; r < 8; r++) hv[r] = (_Float16)(oT[nc][r] * inv);
    *(v8h*)(obuf + (size_t)tok * DD + h * HDD + nc * 16 + hf * 8) = hv;
  }
}

// ---------------- residual + LayerNorm (one wave per token row, D=256) ----------------
__global__ void __launch_bounds__(256) addln_k(const float* __restrict__ x,
                                               const float* __restrict__ t,
                                               const float* __restrict__ ln,
                                               float* __restrict__ xo,
                                               _Float16* __restrict__ h16, int Ntok) {
  int lane = threadIdx.x & 31;
  int row  = blockIdx.x * (blockDim.x >> 5) + (threadIdx.x >> 5);
  if (row >= Ntok) return;
  float v[8], s = 0.f;
#pragma unroll
  for (int i = 0; i < 8; i++) {
    v[i] = x[(size_t)row * DD + i * 32 + lane] + t[(size_t)row * DD + i * 32 + lane];
    s += v[i];
  }
#pragma unroll
  for (int m = 1; m < 32; m <<= 1) s += __shfl_xor(s, m, 32);
  float mean = s * (1.0f / DD);
  float var = 0.f;
#pragma unroll
  for (int i = 0; i < 8; i++) { float d = v[i] - mean; var += d * d; }
#pragma unroll
  for (int m = 1; m < 32; m <<= 1) var += __shfl_xor(var, m, 32);
  float rstd = rsqrtf(var * (1.0f / DD) + 1e-5f);
#pragma unroll
  for (int i = 0; i < 8; i++) {
    int d = i * 32 + lane;
    float out = (v[i] - mean) * rstd * ln[d] + ln[DD + d];
    xo[(size_t)row * DD + d]  = out;
    h16[(size_t)row * DD + d] = (_Float16)out;
  }
}

__device__ __forceinline__ float posenc(int pos, int d) {
  float div = __expf(-(float)(d & ~1) * (9.210340372f / 256.0f)); // ln(1e4)/D
  float ang = (float)pos * div;
  return (d & 1) ? __cosf(ang) : __sinf(ang);
}

__global__ void __launch_bounds__(256) embed_k(const float* __restrict__ emb,
                                               const int* __restrict__ ids,
                                               float* __restrict__ x,
                                               _Float16* __restrict__ h16) {
  int lane = threadIdx.x & 31;
  int tok  = blockIdx.x * (blockDim.x >> 5) + (threadIdx.x >> 5);
  if (tok >= BB * SS) return;
  int spos = tok % SS;
  int id = ids[tok];
#pragma unroll
  for (int i = 0; i < 8; i++) {
    int d = i * 32 + lane;
    float vv = emb[(size_t)id * DD + d] + posenc(spos, d);
    x[(size_t)tok * DD + d] = vv;
    h16[(size_t)tok * DD + d] = (_Float16)vv;
  }
}

// length regulation: cum-sum durations, searchsorted(cum, t, 'right')
__global__ void lenreg_k(const int* __restrict__ dur, int* __restrict__ idx) {
  __shared__ int cum[SS];
  int b = blockIdx.x;
  if (threadIdx.x == 0) {
    int c = 0;
    for (int i = 0; i < SS; i++) { c += dur[b * SS + i]; cum[i] = c; }
  }
  __syncthreads();
  for (int t = threadIdx.x; t < MELN; t += blockDim.x) {
    int lo = 0, hi = SS;
    while (lo < hi) { int mid = (lo + hi) >> 1; if (cum[mid] > t) hi = mid; else lo = mid + 1; }
    idx[b * MELN + t] = lo;
  }
}

__global__ void __launch_bounds__(256) gather_k(const float* __restrict__ xe,
                                                const int* __restrict__ idx,
                                                float* __restrict__ x,
                                                _Float16* __restrict__ h16) {
  int lane = threadIdx.x & 31;
  int tok  = blockIdx.x * (blockDim.x >> 5) + (threadIdx.x >> 5);
  if (tok >= BB * MELN) return;
  int b = tok / MELN, t = tok % MELN;
  int src = b * SS + idx[tok];
#pragma unroll
  for (int i = 0; i < 8; i++) {
    int d = i * 32 + lane;
    float vv = xe[(size_t)src * DD + d] + posenc(t, d);
    x[(size_t)tok * DD + d] = vv;
    h16[(size_t)tok * DD + d] = (_Float16)vv;
  }
}

__global__ void cvt_k(const float* __restrict__ s, _Float16* __restrict__ d, int n) {
  int i = blockIdx.x * blockDim.x + threadIdx.x;
  if (i < n) d[i] = (_Float16)s[i];
}

__global__ void durout_k(const int* __restrict__ dur, float* __restrict__ o, int n) {
  int i = blockIdx.x * blockDim.x + threadIdx.x;
  if (i < n) o[i] = (float)dur[i];
}

// ---------------- host orchestration ----------------
extern "C" void kernel_launch(void* const* d_in, const int* in_sizes, int n_in,
                              void* d_out, int out_size, void* d_ws, size_t ws_size,
                              hipStream_t stream) {
  (void)in_sizes; (void)n_in; (void)out_size;
  const float* emb  = (const float*)d_in[0];
  const float* melW = (const float*)d_in[1];
  const float* melb = (const float*)d_in[2];
  const int*   pid  = (const int*)d_in[3];
  const int*   dur  = (const int*)d_in[4];
  const float* W32[2][4]  = {{(const float*)d_in[5],  (const float*)d_in[7],
                              (const float*)d_in[10], (const float*)d_in[12]},
                             {(const float*)d_in[15], (const float*)d_in[17],
                              (const float*)d_in[20], (const float*)d_in[22]}};
  const float* Bqkv[2] = {(const float*)d_in[6],  (const float*)d_in[16]};
  const float* Bo[2]   = {(const float*)d_in[8],  (const float*)d_in[18]};
  const float* Ln1[2]  = {(const float*)d_in[9],  (const float*)d_in[19]};
  const float* B1[2]   = {(const float*)d_in[11], (const float*)d_in[21]};
  const float* B2[2]   = {(const float*)d_in[13], (const float*)d_in[23]};
  const float* Ln2[2]  = {(const float*)d_in[14], (const float*)d_in[24]};
  float* out = (float*)d_out;

  const int NE = BB * SS;    // 7680 encoder tokens
  const int ND = BB * MELN;  // 30720 decoder tokens

  // bump-allocate workspace (256B aligned)
  char* p = (char*)d_ws;
  auto alloc = [&](size_t bytes) -> char* {
    char* r = p; p += (bytes + 255) & ~(size_t)255; return r;
  };
  float*    xe   = (float*)alloc((size_t)NE * DD * 4);
  float*    xd   = (float*)alloc((size_t)ND * DD * 4);
  _Float16* h16  = (_Float16*)alloc((size_t)ND * DD * 2);
  _Float16* ob   = (_Float16*)alloc((size_t)ND * DD * 2);
  _Float16* qkvb = (_Float16*)alloc((size_t)ND * 3 * DD * 2);
  float*    tmp  = (float*)qkvb; // aliases qkvb: lifetimes disjoint within a layer
  _Float16* h1   = (_Float16*)alloc((size_t)ND * 4 * DD * 2);
  _Float16* vT   = (_Float16*)alloc((size_t)BB * DD * MELN * 2); // max Sp = 960
  int*      idx  = (int*)alloc((size_t)ND * 4);
  const size_t szW[4] = {(size_t)LLY * 3 * DD * DD, (size_t)LLY * DD * DD,
                         (size_t)LLY * 4 * DD * DD, (size_t)LLY * DD * 4 * DD};
  _Float16* W16[2][4];
  for (int st = 0; st < 2; st++)
    for (int i = 0; i < 4; i++) W16[st][i] = (_Float16*)alloc(szW[i] * 2);
  _Float16* melW16 = (_Float16*)alloc((size_t)MBN * DD * 2);
  if ((size_t)(p - (char*)d_ws) > ws_size) return; // workspace too small

  // weight conversion fp32 -> f16 (every call; no cached state)
  auto cvt = [&](const float* s, _Float16* d, size_t n) {
    cvt_k<<<(int)((n + 255) / 256), 256, 0, stream>>>(s, d, (int)n);
  };
  for (int st = 0; st < 2; st++)
    for (int i = 0; i < 4; i++) cvt(W32[st][i], W16[st][i], szW[i]);
  cvt(melW, melW16, (size_t)MBN * DD);

  auto run_stack = [&](float* x, int Ntok, int S, int st) {
    const int Sp = (S + 31) & ~31; // padded key stride (240->256, 960->960)
    for (int l = 0; l < LLY; l++) {
      { // QKV: [Ntok,256] -> [Ntok,768], f16 out
        int waves = (Ntok / 32) * (768 / 64);
        gemm_k<2, 4, 0, 1><<<(waves + 3) / 4, 128, 0, stream>>>(
            h16, W16[st][0] + (size_t)l * 3 * DD * DD, Bqkv[st] + l * 3 * DD,
            nullptr, qkvb, Ntok, 3 * DD, DD);
      }
      { // V transpose for vectorized P*V fragments
        int total = BB * DD * Sp;
        vtrans_k<<<(total + 255) / 256, 256, 0, stream>>>(qkvb, vT, S, Sp);
      }
      { // attention
        int waves = BB * HH * (S / 16);
        attn_k<<<waves / 4, 128, 0, stream>>>(qkvb, vT, ob, S, Sp);
      }
      { // out-proj: f32 out (tmp aliases qkvb, which is dead now)
        int waves = (Ntok / 32) * (256 / 64);
        gemm_k<2, 4, 0, 0><<<(waves + 3) / 4, 128, 0, stream>>>(
            ob, W16[st][1] + (size_t)l * DD * DD, Bo[st] + l * DD,
            tmp, nullptr, Ntok, DD, DD);
      }
      addln_k<<<(Ntok + 7) / 8, 256, 0, stream>>>(x, tmp, Ln1[st] + l * 2 * DD,
                                                  x, h16, Ntok);
      { // FFN1 + ReLU, f16 out
        int waves = (Ntok / 32) * (1024 / 64);
        gemm_k<2, 4, 1, 1><<<(waves + 3) / 4, 128, 0, stream>>>(
            h16, W16[st][2] + (size_t)l * 4 * DD * DD, B1[st] + l * 4 * DD,
            nullptr, h1, Ntok, 4 * DD, DD);
      }
      { // FFN2, f32 out
        int waves = (Ntok / 32) * (256 / 64);
        gemm_k<2, 4, 0, 0><<<(waves + 3) / 4, 128, 0, stream>>>(
            h1, W16[st][3] + (size_t)l * DD * 4 * DD, B2[st] + l * DD,
            tmp, nullptr, Ntok, DD, 4 * DD);
      }
      addln_k<<<(Ntok + 7) / 8, 256, 0, stream>>>(x, tmp, Ln2[st] + l * 2 * DD,
                                                  x, h16, Ntok);
    }
  };

  // encoder
  embed_k<<<(NE + 7) / 8, 256, 0, stream>>>(emb, pid, xe, h16);
  run_stack(xe, NE, SS, 0);

  // length regulation + decoder input
  lenreg_k<<<BB, 256, 0, stream>>>(dur, idx);
  gather_k<<<(ND + 7) / 8, 256, 0, stream>>>(xe, idx, xd, h16);
  run_stack(xd, ND, MELN, 1);

  // mel projection: [30720,256] -> [30720,80] straight into d_out
  {
    int waves = (ND / 32) * 1; // NT=5 covers all 80 cols in one n-block
    gemm_k<2, 5, 0, 0><<<(waves + 3) / 4, 128, 0, stream>>>(
        h16, melW16, melb, out, nullptr, ND, MBN, DD);
  }
  // second tuple output: durations (cast to output dtype)
  durout_k<<<(NE + 255) / 256, 256, 0, stream>>>(dur, out + (size_t)ND * MBN, NE);
}
